// RecursiveEncoder_16681652978506
// MI455X (gfx1250) — matrix operand order repaired and checked by reading, not compiled, not run on hardware
//
#include <hip/hip_runtime.h>
#include <hip/hip_bf16.h>

typedef __attribute__((ext_vector_type(16))) _Float16 v16h;
typedef __attribute__((ext_vector_type(8)))  _Float16 v8h;
typedef __attribute__((ext_vector_type(8)))  float    v8f;

#define B_   4096
#define M_   10
#define E_   40
#define T_   4

#define KC_C   10          // 320/32 K-chunks for child GEMM
#define KC_I   17          // 544/32 (516 padded) for edge GEMMs
#define KC_P   24          // 768/32 for parent GEMM
#define KPAD_C 320
#define KPAD_I 544
#define KPAD_P 768
#define NT_    16          // 256 / 16 N-tiles
#define FRAG_  512         // f16 elements per packed B fragment (32 lanes * 16)

// ---------------- WMMA helpers ----------------

static __device__ __forceinline__ v8f wmma_f16(v16h a, v16h b, v8f c) {
  // D = A(16x32 f16) x B(32x16 f16) + C(16x16 f32)
  return __builtin_amdgcn_wmma_f32_16x16x32_f16(
      /*neg_a=*/false, a, /*neg_b=*/false, b,
      /*c_mod=*/(short)0, c, /*reuse_a=*/false, /*reuse_b=*/false);
}

// A-matrix 16x32 f16 per-ISA layout: lane holds row M=lane&15;
// lanes 0-15: K = {0..7, 16..23}; lanes 16-31: K = {8..15, 24..31}.
// rowp = &A[row * stride] (f16 units, 16B-aligned rows).
static __device__ __forceinline__ v16h load_a_frag(const _Float16* rowp, int kc, int hi) {
  union { v16h v; v8h h[2]; } u;
  const _Float16* p = rowp + kc * 32 + hi * 8;
  u.h[0] = *(const v8h*)(p);        // K = kc*32 + hi*8 + 0..7
  u.h[1] = *(const v8h*)(p + 16);   // K = kc*32 + 16 + hi*8 + 0..7
  return u.v;
}

// B fragments pre-packed so each lane's 16 f16 are contiguous (32B load).
static __device__ __forceinline__ v16h load_b_frag(const _Float16* WP, int kc, int nt, int lane) {
  return *(const v16h*)(WP + ((size_t)(kc * NT_ + nt)) * FRAG_ + (size_t)lane * 16);
}

// ---------------- weight pack kernel (f32 -> f16, B-fragment layout) ----------------

__global__ void pack_w_kernel(const float* __restrict__ Wc, const float* __restrict__ Wi0,
                              const float* __restrict__ Wi1, const float* __restrict__ Wp,
                              _Float16* __restrict__ out) {
  const int szC = KC_C * NT_ * FRAG_;
  const int szI = KC_I * NT_ * FRAG_;
  const int szP = KC_P * NT_ * FRAG_;
  const int total = szC + 2 * szI + szP;
  int idx = blockIdx.x * blockDim.x + threadIdx.x;
  if (idx >= total) return;

  const float* W; int kreal; int local;
  if (idx < szC)            { W = Wc;  kreal = 320; local = idx; }
  else if (idx < szC + szI) { W = Wi0; kreal = 516; local = idx - szC; }
  else if (idx < szC + 2*szI) { W = Wi1; kreal = 516; local = idx - szC - szI; }
  else                      { W = Wp;  kreal = 768; local = idx - szC - 2*szI; }

  int kc   = local / (NT_ * FRAG_);
  int rem  = local % (NT_ * FRAG_);
  int nt   = rem / FRAG_;
  int lrem = rem % FRAG_;
  int lane = lrem / 16;
  int j    = lrem % 16;
  int n = nt * 16 + (lane & 15);
  int k = kc * 32 + (lane >> 4) * 8 + (j & 7) + (j >> 3) * 16;     // lanes 0-15: K {0..7,16..23}; 16-31: K {8..15,24..31}
  float v = (k < kreal) ? W[(size_t)k * 256 + n] : 0.f;
  out[idx] = (_Float16)v;
}

// ---------------- main encoder kernel: one block per batch element ----------------

__global__ __launch_bounds__(256) void encoder_kernel(
    const float* __restrict__ child_feats, const float* __restrict__ child_exists,
    const float* __restrict__ e1h, const int* __restrict__ eidx,
    const float* __restrict__ bc, const float* __restrict__ bi0,
    const float* __restrict__ bi1, const float* __restrict__ bp,
    const _Float16* __restrict__ WcP, const _Float16* __restrict__ Wi0P,
    const _Float16* __restrict__ Wi1P, const _Float16* __restrict__ WpP,
    float* __restrict__ out) {
  __shared__ __align__(16) _Float16 Af[48 * KPAD_I];   // reused: 16x320 / 48x544 / 16x768
  __shared__ __align__(16) _Float16 cfh[16 * 256];     // current node features (f16)
  __shared__ float acc[16 * 256];                      // segment-sum accumulator
  __shared__ float feats[768];                         // 3 column-max feature blocks
  __shared__ float cnt[16];
  __shared__ float rowmask[16];
  __shared__ int   sidx[E_], didx[E_];

  const int b    = blockIdx.x;
  const int tid  = threadIdx.x;
  const int lane = tid & 31;
  const int wave = tid >> 5;      // 0..7
  const int col  = lane & 15;
  const int hi   = lane >> 4;
  const int nt0  = wave;          // N-tiles handled by this wave
  const int nt1  = wave + 8;

  // ---- init: feats=0, row mask, edge indices, A = f16(child_feats) 16x320 ----
  for (int i = tid; i < 768; i += 256) feats[i] = 0.f;
  if (tid < 16) rowmask[tid] = (tid < M_) ? child_exists[(size_t)b * M_ + tid] : 0.f;
  if (tid < E_) {
    sidx[tid] = eidx[((size_t)b * E_ + tid) * 2 + 0];
    didx[tid] = eidx[((size_t)b * E_ + tid) * 2 + 1];
  }
  for (int i = tid; i < 16 * (KPAD_C / 2); i += 256) {
    int row = i / (KPAD_C / 2);
    int c2  = i % (KPAD_C / 2);
    union { _Float16 h[2]; unsigned u; } p;
    if (row < M_) {
      const float* src = child_feats + ((size_t)b * M_ + row) * KPAD_C + c2 * 2;
      p.h[0] = (_Float16)src[0];
      p.h[1] = (_Float16)src[1];
    } else {
      p.u = 0u;
    }
    *(unsigned*)&Af[row * KPAD_C + c2 * 2] = p.u;
  }
  __syncthreads();

  // ---- phase 1: cf = relu(A @ Wc + bc) * mask ; feats[0:256] = colmax ----
  {
    v8f c0 = {}, c1 = {};
    const _Float16* arow = Af + col * KPAD_C;
    for (int kc = 0; kc < KC_C; ++kc) {
      v16h bf0 = load_b_frag(WcP, kc, nt0, lane);
      v16h bf1 = load_b_frag(WcP, kc, nt1, lane);
      v16h a   = load_a_frag(arow, kc, hi);
      c0 = wmma_f16(a, bf0, c0);
      c1 = wmma_f16(a, bf1, c1);
    }
    union { v8f v; float f[8]; } u0, u1;
    u0.v = c0; u1.v = c1;
    const int n0 = nt0 * 16 + col, n1 = nt1 * 16 + col;
    const float bb0 = bc[n0], bb1 = bc[n1];
    float mx0 = 0.f, mx1 = 0.f;
#pragma unroll
    for (int r = 0; r < 8; ++r) {
      int m = hi * 8 + r;                 // D layout: vgpr r, lanes>=16 -> M=8+r
      float v0 = fmaxf(u0.f[r] + bb0, 0.f) * rowmask[m];
      float v1 = fmaxf(u1.f[r] + bb1, 0.f) * rowmask[m];
      cfh[m * 256 + n0] = (_Float16)v0;
      cfh[m * 256 + n1] = (_Float16)v1;
      mx0 = fmaxf(mx0, v0);
      mx1 = fmaxf(mx1, v1);
    }
    atomicMax((int*)&feats[n0], __float_as_int(mx0));  // all values >= 0
    atomicMax((int*)&feats[n1], __float_as_int(mx1));
  }

  // ---- phase 2: two message-passing iterations ----
  for (int it = 0; it < 2; ++it) {
    const _Float16* WiP = it ? Wi1P : Wi0P;
    const float*    bi  = it ? bi1  : bi0;
    __syncthreads();                       // cfh ready / acc free
    for (int i = tid; i < 16 * 256; i += 256) acc[i] = 0.f;
    if (tid < 16) cnt[tid] = 0.f;
    // build Ane: 48 x 544, row e = [cf[src], cf[dst], onehot, 0-pad]
    for (int i = tid; i < 48 * (KPAD_I / 2); i += 256) {
      int row = i / (KPAD_I / 2);
      int c2  = i % (KPAD_I / 2);
      int c   = c2 * 2;
      unsigned u = 0u;
      if (row < E_) {
        if (c < 256) {
          u = *(const unsigned*)&cfh[sidx[row] * 256 + c];
        } else if (c < 512) {
          u = *(const unsigned*)&cfh[didx[row] * 256 + (c - 256)];
        } else if (c < 512 + T_) {
          union { _Float16 h[2]; unsigned uu; } p;
          const float* ep = e1h + ((size_t)b * E_ + row) * T_ + (c - 512);
          p.h[0] = (_Float16)ep[0];
          p.h[1] = (_Float16)ep[1];
          u = p.uu;
        }
      }
      *(unsigned*)&Af[row * KPAD_I + c] = u;
    }
    __syncthreads();
    // GEMM: (48x544) @ (544x256), 3 M-tiles x 2 N-tiles per wave
    v8f vc[3][2];
#pragma unroll
    for (int mt = 0; mt < 3; ++mt) { vc[mt][0] = (v8f){}; vc[mt][1] = (v8f){}; }
    for (int kc = 0; kc < KC_I; ++kc) {
      v16h bf0 = load_b_frag(WiP, kc, nt0, lane);
      v16h bf1 = load_b_frag(WiP, kc, nt1, lane);
#pragma unroll
      for (int mt = 0; mt < 3; ++mt) {
        v16h a = load_a_frag(Af + (mt * 16 + col) * KPAD_I, kc, hi);
        vc[mt][0] = wmma_f16(a, bf0, vc[mt][0]);
        vc[mt][1] = wmma_f16(a, bf1, vc[mt][1]);
      }
    }
    if (tid < E_) atomicAdd(&cnt[sidx[tid]], 1.f);
    // relu + bias, scatter-add into segment accumulator
#pragma unroll
    for (int t = 0; t < 2; ++t) {
      int nt = t ? nt1 : nt0;
      int n  = nt * 16 + col;
      float bias = bi[n];
#pragma unroll
      for (int mt = 0; mt < 3; ++mt) {
        union { v8f v; float f[8]; } uc;
        uc.v = vc[mt][t];
#pragma unroll
        for (int r = 0; r < 8; ++r) {
          int e = mt * 16 + hi * 8 + r;
          if (e < E_) {
            float v = fmaxf(uc.f[r] + bias, 0.f);
            atomicAdd(&acc[sidx[e] * 256 + n], v);
          }
        }
      }
    }
    __syncthreads();
    // cf = acc / max(cnt,1); feats[(it+1)*256 + n] = colmax
    for (int i = tid; i < 16 * 256; i += 256) {
      int m = i >> 8, n = i & 255;
      float v = (m < M_) ? acc[i] / fmaxf(cnt[m], 1.f) : 0.f;
      cfh[i] = (_Float16)v;
      atomicMax((int*)&feats[(it + 1) * 256 + n], __float_as_int(v));
    }
  }
  __syncthreads();

  // ---- phase 3: out = relu(feats(1x768) @ Wp + bp) ----
  for (int i = tid; i < 16 * (KPAD_P / 2); i += 256) {
    int row = i / (KPAD_P / 2);
    int c2  = i % (KPAD_P / 2);
    union { _Float16 h[2]; unsigned u; } p;
    if (row == 0) {
      p.h[0] = (_Float16)feats[c2 * 2];
      p.h[1] = (_Float16)feats[c2 * 2 + 1];
    } else {
      p.u = 0u;
    }
    *(unsigned*)&Af[row * KPAD_P + c2 * 2] = p.u;
  }
  __syncthreads();
  {
    v8f c0 = {}, c1 = {};
    const _Float16* arow = Af + col * KPAD_P;
    for (int kc = 0; kc < KC_P; ++kc) {
      v16h bf0 = load_b_frag(WpP, kc, nt0, lane);
      v16h bf1 = load_b_frag(WpP, kc, nt1, lane);
      v16h a   = load_a_frag(arow, kc, hi);
      c0 = wmma_f16(a, bf0, c0);
      c1 = wmma_f16(a, bf1, c1);
    }
    if (hi == 0) {                       // M=0 lives in vgpr 0 of lanes 0..15
      union { v8f v; float f[8]; } u0, u1;
      u0.v = c0; u1.v = c1;
      int n0 = nt0 * 16 + col, n1 = nt1 * 16 + col;
      out[(size_t)b * 256 + n0] = fmaxf(u0.f[0] + bp[n0], 0.f);
      out[(size_t)b * 256 + n1] = fmaxf(u1.f[0] + bp[n1], 0.f);
    }
  }
}

extern "C" void kernel_launch(void* const* d_in, const int* in_sizes, int n_in,
                              void* d_out, int out_size, void* d_ws, size_t ws_size,
                              hipStream_t stream) {
  (void)in_sizes; (void)n_in; (void)out_size; (void)ws_size;
  const float* child_feats  = (const float*)d_in[0];
  const float* child_exists = (const float*)d_in[1];
  const float* e1h          = (const float*)d_in[2];
  const int*   eidx         = (const int*)d_in[3];
  const float* Wc  = (const float*)d_in[4];
  const float* bc  = (const float*)d_in[5];
  const float* Wi0 = (const float*)d_in[6];
  const float* bi0 = (const float*)d_in[7];
  const float* Wi1 = (const float*)d_in[8];
  const float* bi1 = (const float*)d_in[9];
  const float* Wp  = (const float*)d_in[10];
  const float* bp  = (const float*)d_in[11];

  _Float16* ws = (_Float16*)d_ws;
  const int szC = KC_C * NT_ * FRAG_;
  const int szI = KC_I * NT_ * FRAG_;
  const int szP = KC_P * NT_ * FRAG_;
  _Float16* WcP  = ws;
  _Float16* Wi0P = WcP + szC;
  _Float16* Wi1P = Wi0P + szI;
  _Float16* WpP  = Wi1P + szI;

  const int total = szC + 2 * szI + szP;
  pack_w_kernel<<<(total + 255) / 256, 256, 0, stream>>>(Wc, Wi0, Wi1, Wp, ws);
  encoder_kernel<<<B_, 256, 0, stream>>>(child_feats, child_exists, e1h, eidx,
                                         bc, bi0, bi1, bp,
                                         WcP, Wi0P, Wi1P, WpP, (float*)d_out);
}
